// MultiHeadAttention_3710851744507
// MI455X (gfx1250) — matrix operand order repaired and checked by reading, not compiled, not run on hardware
//
#include <hip/hip_runtime.h>
#include <hip/hip_bf16.h>

// ---- problem constants (match reference) ----
#define BB 2
#define SS 2048
#define DD 1024
#define HH 16
#define DKK 64
#define SC_STRIDE 2052   // 2048 + 4 floats pad -> bank offset 4 per row, conflict free
#define KROW 72          // K-stage LDS row stride in halves (144 B: conflict-free, 16B aligned)

typedef __attribute__((ext_vector_type(16))) _Float16 v16h;
typedef __attribute__((ext_vector_type(8)))  _Float16 v8h;
typedef __attribute__((ext_vector_type(8)))  float    v8f;

#define DEV __device__ __forceinline__

// ---- CDNA5 async global->LDS copy path (ASYNCcnt tracked) ----------------
#if defined(__has_builtin)
#if __has_builtin(__builtin_amdgcn_global_load_async_to_lds_b128)
#define HAS_ASYNC_COPY 1
#endif
#endif
#ifndef HAS_ASYNC_COPY
#define HAS_ASYNC_COPY 0
#endif

#if HAS_ASYNC_COPY
typedef __attribute__((__vector_size__(4 * sizeof(int)))) int v4i;
typedef __attribute__((address_space(1))) v4i* g_v4i_p;   // global int4*
typedef __attribute__((address_space(3))) v4i* l_v4i_p;   // LDS    int4*

DEV void wait_async0() {
#if __has_builtin(__builtin_amdgcn_s_wait_asynccnt)
  __builtin_amdgcn_s_wait_asynccnt(0);
#else
  asm volatile("s_wait_asynccnt 0" ::: "memory");
#endif
}
// One call copies 16B/lane; four calls move a 16x64-half K tile (2 KB).
// INST_OFFSET applies to both global and LDS addresses (ISA async semantics).
DEV void stage_k_tile(const _Float16* g, _Float16* l) {
  g_v4i_p gp = (g_v4i_p)const_cast<_Float16*>(g);
  l_v4i_p lp = (l_v4i_p)l;
  __builtin_amdgcn_global_load_async_to_lds_b128(gp, lp, 0, 0);
  __builtin_amdgcn_global_load_async_to_lds_b128(gp, lp, 32, 0);
  __builtin_amdgcn_global_load_async_to_lds_b128(gp, lp, 64, 0);
  __builtin_amdgcn_global_load_async_to_lds_b128(gp, lp, 96, 0);
}
#endif

// ---------------- WMMA helpers (CDNA5 16x16x32 f16 -> f32) ----------------
DEV v8f wmma32(v16h a, v16h b, v8f c) {
  return __builtin_amdgcn_wmma_f32_16x16x32_f16(false, a, false, b, (short)0, c,
                                                false, false);
}

// A fragment in this file's contiguous k order: halves a[0..7] = row[k0+hi*16 ..+7],
// a[8..15] = row[k0+hi*16+8 ..+7] (two contiguous b128 runs).
DEV v16h a_frag_from_f16(const _Float16* __restrict__ row, int k0, int hi) {
  v8h lo = *reinterpret_cast<const v8h*>(row + k0 + hi * 16);
  v8h hh = *reinterpret_cast<const v8h*>(row + k0 + hi * 16 + 8);
  v16h a;
#pragma unroll
  for (int t = 0; t < 8; ++t) { a[t] = lo[t]; a[8 + t] = hh[t]; }
  return a;
}

DEV v16h a_frag_from_f32(const float* __restrict__ row, int k0, int hi) {
  float4 f0 = *reinterpret_cast<const float4*>(row + k0 + hi * 16);
  float4 f1 = *reinterpret_cast<const float4*>(row + k0 + hi * 16 + 4);
  float4 f2 = *reinterpret_cast<const float4*>(row + k0 + hi * 16 + 8);
  float4 f3 = *reinterpret_cast<const float4*>(row + k0 + hi * 16 + 12);
  v16h a;
  a[0]=(_Float16)f0.x; a[1]=(_Float16)f0.y; a[2]=(_Float16)f0.z; a[3]=(_Float16)f0.w;
  a[4]=(_Float16)f1.x; a[5]=(_Float16)f1.y; a[6]=(_Float16)f1.z; a[7]=(_Float16)f1.w;
  a[8]=(_Float16)f2.x; a[9]=(_Float16)f2.y; a[10]=(_Float16)f2.z; a[11]=(_Float16)f2.w;
  a[12]=(_Float16)f3.x; a[13]=(_Float16)f3.y; a[14]=(_Float16)f3.z; a[15]=(_Float16)f3.w;
  return a;
}

// A fragment from LDS f32 scores; 8-runs are uniformly active/masked
// (nact multiple of 16, runs 8-aligned).
DEV v16h a_frag_from_lds(const float* __restrict__ srow, int c0, int nact) {
  v16h a;
  if (c0 < nact) {
    float4 f0 = *reinterpret_cast<const float4*>(srow + c0);
    float4 f1 = *reinterpret_cast<const float4*>(srow + c0 + 4);
    a[0]=(_Float16)f0.x; a[1]=(_Float16)f0.y; a[2]=(_Float16)f0.z; a[3]=(_Float16)f0.w;
    a[4]=(_Float16)f1.x; a[5]=(_Float16)f1.y; a[6]=(_Float16)f1.z; a[7]=(_Float16)f1.w;
  } else {
#pragma unroll
    for (int t = 0; t < 8; ++t) a[t] = (_Float16)0.0f;
  }
  int c1 = c0 + 8;
  if (c1 < nact) {
    float4 f2 = *reinterpret_cast<const float4*>(srow + c1);
    float4 f3 = *reinterpret_cast<const float4*>(srow + c1 + 4);
    a[8]=(_Float16)f2.x; a[9]=(_Float16)f2.y; a[10]=(_Float16)f2.z; a[11]=(_Float16)f2.w;
    a[12]=(_Float16)f3.x; a[13]=(_Float16)f3.y; a[14]=(_Float16)f3.z; a[15]=(_Float16)f3.w;
  } else {
#pragma unroll
    for (int t = 8; t < 16; ++t) a[t] = (_Float16)0.0f;
  }
  return a;
}

// B fragment: lane n holds 16 contiguous K-halves of column n.
DEV v16h b_frag(const _Float16* __restrict__ p) {         // 32B-aligned source
  return *reinterpret_cast<const v16h*>(p);
}
DEV v16h b_frag16(const _Float16* __restrict__ p) {       // 16B-aligned (LDS) source
  v8h lo = *reinterpret_cast<const v8h*>(p);
  v8h hh = *reinterpret_cast<const v8h*>(p + 8);
  v16h b;
#pragma unroll
  for (int t = 0; t < 8; ++t) { b[t] = lo[t]; b[8 + t] = hh[t]; }
  return b;
}

// ---------------- Kernel 0: weight transpose + f32->f16 convert -----------
__global__ void wt_kernel(const float* __restrict__ w, _Float16* __restrict__ wt) {
  int idx = blockIdx.x * blockDim.x + threadIdx.x;
  if (idx >= DD * DD) return;
  int n = idx >> 10;
  int k = idx & (DD - 1);
  wt[idx] = (_Float16)w[(size_t)k * DD + n];
}

// ---------------- Kernel 1: projections X(f32) @ W -> head-split f16 ------
// Wave: 16x64 strip, 4 accumulators. Copy-free ping-pong pipeline:
// two named fragment sets, loads issue one 32-step ahead of their WMMAs.
__global__ __launch_bounds__(128)
void proj_kernel(const float* __restrict__ X, const _Float16* __restrict__ Wt,
                 _Float16* __restrict__ dst, int mode) {
  int w    = threadIdx.x >> 5;
  int lane = threadIdx.x & 31;
  int m  = lane & 15;
  int hi = lane >> 4;
  int rt = blockIdx.y;
  int cg = blockIdx.x * 4 + w;
  const float*    arow = X  + (size_t)(rt * 16 + m) * DD;
  const _Float16* wtb  = Wt + (size_t)(cg * 64 + m) * DD;
  v8f acc[4] = {};
  v16h a0 = a_frag_from_f32(arow, 0, hi);
  v16h a1;
  v16h b0[4], b1[4];
#pragma unroll
  for (int nt = 0; nt < 4; ++nt) b0[nt] = b_frag(wtb + (size_t)nt * 16 * DD + hi * 16);
  for (int k0 = 0; k0 < DD; k0 += 64) {       // DD % 64 == 0
    a1 = a_frag_from_f32(arow, k0 + 32, hi);  // load set1 (k0+32)
#pragma unroll
    for (int nt = 0; nt < 4; ++nt)
      b1[nt] = b_frag(wtb + (size_t)nt * 16 * DD + k0 + 32 + hi * 16);
#pragma unroll
    for (int nt = 0; nt < 4; ++nt) acc[nt] = wmma32(a0, b0[nt], acc[nt]);
    if (k0 + 64 < DD) {                       // load set0 (k0+64)
      a0 = a_frag_from_f32(arow, k0 + 64, hi);
#pragma unroll
      for (int nt = 0; nt < 4; ++nt)
        b0[nt] = b_frag(wtb + (size_t)nt * 16 * DD + k0 + 64 + hi * 16);
    }
#pragma unroll
    for (int nt = 0; nt < 4; ++nt) acc[nt] = wmma32(a1, b1[nt], acc[nt]);
  }
#pragma unroll
  for (int nt = 0; nt < 4; ++nt) {
#pragma unroll
    for (int r = 0; r < 8; ++r) {
      int orow = rt * 16 + r + 8 * hi;
      int ocol = cg * 64 + nt * 16 + m;
      int b_ = orow >> 11;
      int s  = orow & (SS - 1);
      int h  = ocol >> 6;
      int dk = ocol & (DKK - 1);
      int bh = b_ * HH + h;
      _Float16 v = (_Float16)acc[nt][r];
      if (mode == 0) dst[((size_t)bh * SS + s) * DKK + dk] = v;
      else           dst[((size_t)bh * DKK + dk) * SS + s] = v;
    }
  }
}

// ---------------- Kernel 2: causal attention per (b,h,16-row q-tile) ------
__global__ __launch_bounds__(128)
void attn_kernel(const _Float16* __restrict__ qh, const _Float16* __restrict__ kh,
                 const _Float16* __restrict__ vT, float* __restrict__ attn_out,
                 _Float16* __restrict__ ctx) {
  extern __shared__ float sc[];                 // 16 x SC_STRIDE f32
  __shared__ float red[16][8];
  __shared__ float rmax[16];
  __shared__ float rinv[16];
#if HAS_ASYNC_COPY
  __shared__ _Float16 kstage[4][2][16 * KROW];  // per-wave double-buffered K tiles
#endif

  int tid  = threadIdx.x;
  int w    = tid >> 5;
  int lane = tid & 31;
  int m    = lane & 15;
  int hi   = lane >> 4;

  int idx = blockIdx.x;
  int qt  = idx & 127;
  int bh  = idx >> 7;
  int q0  = qt * 16;
  int nkt = qt + 1;
  int nact = nkt * 16;

  const _Float16* qbase = qh + (size_t)bh * SS * DKK;
  const _Float16* kbase = kh + (size_t)bh * SS * DKK;
  const _Float16* vbase = vT + (size_t)bh * DKK * SS;

  const _Float16* qrow = qbase + (size_t)(q0 + m) * DKK;
  v16h aq0 = a_frag_from_f16(qrow, 0, hi);
  v16h aq1 = a_frag_from_f16(qrow, 32, hi);

  // ---- scores: wave w handles key tiles w, w+4, ...; async DMA staging ----
#if HAS_ASYNC_COPY
  // lane -> (row r_ = lane>>1, 16B chunk c_ = lane&1); 4 async b128 ops / tile
  int r_ = lane >> 1, c_ = lane & 1;
  if (w < nkt)
    stage_k_tile(kbase + (size_t)(w * 16 + r_) * DKK + c_ * 8,
                 &kstage[w][0][r_ * KROW + c_ * 8]);
  int bufi = 0;
  for (int kt = w; kt < nkt; kt += 4) {
    wait_async0();                              // tile kt resident in LDS
    if (kt + 4 < nkt)                           // overlap next tile's DMA
      stage_k_tile(kbase + (size_t)((kt + 4) * 16 + r_) * DKK + c_ * 8,
                   &kstage[w][bufi ^ 1][r_ * KROW + c_ * 8]);
    const _Float16* krow = &kstage[w][bufi][m * KROW];
    bufi ^= 1;
    v8f acc = {};
    acc = wmma32(aq0, b_frag16(krow + hi * 16), acc);
    acc = wmma32(aq1, b_frag16(krow + 32 + hi * 16), acc);
    int j = kt * 16 + m;
#pragma unroll
    for (int r = 0; r < 8; ++r) {
      int i_loc = r + 8 * hi;
      int i = q0 + i_loc;
      float v = acc[r] * 0.125f;
      if (j > i) v -= 1e9f;
      sc[i_loc * SC_STRIDE + j] = v;
    }
  }
#else
  for (int kt = w; kt < nkt; kt += 4) {
    const _Float16* krow = kbase + (size_t)(kt * 16 + m) * DKK;
    if (kt + 4 < nkt)
      __builtin_prefetch(kbase + (size_t)((kt + 4) * 16 + m) * DKK, 0, 1);
    v8f acc = {};
    acc = wmma32(aq0, b_frag(krow + hi * 16), acc);
    acc = wmma32(aq1, b_frag(krow + 32 + hi * 16), acc);
    int j = kt * 16 + m;
#pragma unroll
    for (int r = 0; r < 8; ++r) {
      int i_loc = r + 8 * hi;
      int i = q0 + i_loc;
      float v = acc[r] * 0.125f;
      if (j > i) v -= 1e9f;
      sc[i_loc * SC_STRIDE + j] = v;
    }
  }
#endif
  __syncthreads();

  // ---- softmax over active prefix: 8 threads per row ----
  int row = tid >> 3;
  int lr  = tid & 7;
  float mx = -3.0e38f;
  for (int c = lr; c < nact; c += 8) mx = fmaxf(mx, sc[row * SC_STRIDE + c]);
  red[row][lr] = mx;
  __syncthreads();
  if (lr == 0) {
    float v = red[row][0];
#pragma unroll
    for (int t = 1; t < 8; ++t) v = fmaxf(v, red[row][t]);
    rmax[row] = v;
  }
  __syncthreads();
  float rm = rmax[row];
  float sum = 0.0f;
  for (int c = lr; c < nact; c += 8) {
    float e = __expf(sc[row * SC_STRIDE + c] - rm);
    sc[row * SC_STRIDE + c] = e;
    sum += e;
  }
  red[row][lr] = sum;
  __syncthreads();
  if (lr == 0) {
    float v = 0.0f;
#pragma unroll
    for (int t = 0; t < 8; ++t) v += red[row][t];
    rinv[row] = 1.0f / v;
  }
  __syncthreads();

  // ---- write normalized attention weights, float4 coalesced bursts ----
  size_t abase = ((size_t)bh * SS + q0) * SS;
  for (int r = 0; r < 16; ++r) {
    float inv = rinv[r];
    for (int c = tid * 4; c < SS; c += 512) {
      float4 p;
      if (c < nact) {
        float4 e = *reinterpret_cast<const float4*>(sc + r * SC_STRIDE + c);
        p.x = e.x * inv; p.y = e.y * inv; p.z = e.z * inv; p.w = e.w * inv;
      } else {
        p.x = 0.0f; p.y = 0.0f; p.z = 0.0f; p.w = 0.0f;
      }
      *reinterpret_cast<float4*>(attn_out + abase + (size_t)r * SS + c) = p;
    }
  }

  // ---- context = P @ V : wave w owns dk columns [16w, 16w+16) ----
  // Copy-free 2-way unrolled pipeline over 32-wide key steps.
  int nks = (nkt + 1) >> 1;
  v8f cacc = {};
  const float* srow = sc + m * SC_STRIDE;
  const _Float16* vb = vbase + (size_t)(w * 16 + m) * SS;
  v16h bv0 = b_frag(vb + hi * 16);              // ks = 0
  int ks = 0;
  for (; ks + 2 <= nks; ks += 2) {
    v16h bv1 = b_frag(vb + (ks + 1) * 32 + hi * 16);
    v16h a0 = a_frag_from_lds(srow, ks * 32 + hi * 16, nact);
    cacc = wmma32(a0, bv0, cacc);
    if (ks + 2 < nks) bv0 = b_frag(vb + (ks + 2) * 32 + hi * 16);
    v16h a1 = a_frag_from_lds(srow, (ks + 1) * 32 + hi * 16, nact);
    cacc = wmma32(a1, bv1, cacc);
  }
  if (ks < nks) {                               // odd tail
    v16h a0 = a_frag_from_lds(srow, ks * 32 + hi * 16, nact);
    cacc = wmma32(a0, bv0, cacc);
  }
  int b_ = bh >> 4;
  int h_ = bh & 15;
#pragma unroll
  for (int r = 0; r < 8; ++r) {
    int i_loc = r + 8 * hi;
    int i = q0 + i_loc;
    float v = cacc[r] * rinv[i_loc];
    ctx[((size_t)(b_ * SS + i)) * DD + h_ * DKK + w * 16 + m] = (_Float16)v;
  }
}

// ---------------- Kernel 3: output = ctx(f16) @ wo -> f32 -----------------
__global__ __launch_bounds__(128)
void out_gemm_kernel(const _Float16* __restrict__ A, const _Float16* __restrict__ Wt,
                     float* __restrict__ out) {
  int w    = threadIdx.x >> 5;
  int lane = threadIdx.x & 31;
  int m  = lane & 15;
  int hi = lane >> 4;
  int rt = blockIdx.y;
  int cg = blockIdx.x * 4 + w;
  const _Float16* arow = A  + (size_t)(rt * 16 + m) * DD;
  const _Float16* wtb  = Wt + (size_t)(cg * 64 + m) * DD;
  v8f acc[4] = {};
  v16h a0 = a_frag_from_f16(arow, 0, hi);
  v16h a1;
  v16h b0[4], b1[4];
#pragma unroll
  for (int nt = 0; nt < 4; ++nt) b0[nt] = b_frag(wtb + (size_t)nt * 16 * DD + hi * 16);
  for (int k0 = 0; k0 < DD; k0 += 64) {
    a1 = a_frag_from_f16(arow, k0 + 32, hi);
#pragma unroll
    for (int nt = 0; nt < 4; ++nt)
      b1[nt] = b_frag(wtb + (size_t)nt * 16 * DD + k0 + 32 + hi * 16);
#pragma unroll
    for (int nt = 0; nt < 4; ++nt) acc[nt] = wmma32(a0, b0[nt], acc[nt]);
    if (k0 + 64 < DD) {
      a0 = a_frag_from_f16(arow, k0 + 64, hi);
#pragma unroll
      for (int nt = 0; nt < 4; ++nt)
        b0[nt] = b_frag(wtb + (size_t)nt * 16 * DD + k0 + 64 + hi * 16);
    }
#pragma unroll
    for (int nt = 0; nt < 4; ++nt) acc[nt] = wmma32(a1, b1[nt], acc[nt]);
  }
#pragma unroll
  for (int nt = 0; nt < 4; ++nt) {
#pragma unroll
    for (int r = 0; r < 8; ++r) {
      int orow = rt * 16 + r + 8 * hi;
      int ocol = cg * 64 + nt * 16 + m;
      out[(size_t)orow * DD + ocol] = acc[nt][r];
    }
  }
}

// ---------------- host-side launch ----------------------------------------
extern "C" void kernel_launch(void* const* d_in, const int* in_sizes, int n_in,
                              void* d_out, int out_size, void* d_ws, size_t ws_size,
                              hipStream_t stream) {
  const float* query = (const float*)d_in[0];
  const float* key_  = (const float*)d_in[1];
  const float* value = (const float*)d_in[2];
  const float* wq = (const float*)d_in[4];
  const float* wk = (const float*)d_in[5];
  const float* wv = (const float*)d_in[6];
  const float* wo = (const float*)d_in[7];

  float* out  = (float*)d_out;                          // (B,S,D) f32
  float* attn = out + (size_t)BB * SS * DD;             // (B,H,S,S) f32

  char* ws = (char*)d_ws;
  _Float16* wqT = (_Float16*)(ws + (0ull  << 20));
  _Float16* wkT = (_Float16*)(ws + (2ull  << 20));
  _Float16* wvT = (_Float16*)(ws + (4ull  << 20));
  _Float16* woT = (_Float16*)(ws + (6ull  << 20));
  _Float16* qh  = (_Float16*)(ws + (8ull  << 20));      // (B,H,S,DK) f16
  _Float16* kh  = (_Float16*)(ws + (16ull << 20));      // (B,H,S,DK) f16
  _Float16* vT  = (_Float16*)(ws + (24ull << 20));      // (B,H,DK,S) f16
  _Float16* ctx = (_Float16*)(ws + (32ull << 20));      // (B*S,D)   f16

  int ntr = (DD * DD + 255) / 256;
  wt_kernel<<<ntr, 256, 0, stream>>>(wq, wqT);
  wt_kernel<<<ntr, 256, 0, stream>>>(wk, wkT);
  wt_kernel<<<ntr, 256, 0, stream>>>(wv, wvT);
  wt_kernel<<<ntr, 256, 0, stream>>>(wo, woT);

  dim3 pg(4, 256), pb(128);
  proj_kernel<<<pg, pb, 0, stream>>>(query, wqT, qh, 0);
  proj_kernel<<<pg, pb, 0, stream>>>(key_,  wkT, kh, 0);
  proj_kernel<<<pg, pb, 0, stream>>>(value, wvT, vT, 1);

  size_t lds = (size_t)16 * SC_STRIDE * sizeof(float);
  attn_kernel<<<BB * HH * (SS / 16), 128, lds, stream>>>(qh, kh, vT, attn, ctx);

  out_gemm_kernel<<<pg, pb, 0, stream>>>(ctx, woT, out);
}
